// AttentionHead_58377195487670
// MI455X (gfx1250) — matrix operand order repaired and checked by reading, not compiled, not run on hardware
//
#include <hip/hip_runtime.h>
#include <hip/hip_bf16.h>
#include <math.h>

// ---------------------------------------------------------------------------
// AttentionHead: B=4, T=4096, C=512 (fp32 in/out)
//   q = toks @ Wq.T ; k = toks @ Wk.T ; v = toks @ Wv.T   (bf16 to workspace)
//   out = softmax(causal(q k^T / sqrt(C))) v               (flash, online softmax)
// Workspace (bf16 bits as ushort): qw [B*T][C] | kT [B][C][T] | vw [B*T][C]
// ---------------------------------------------------------------------------

#define BATCH 4
#define TSEQ  4096
#define CDIM  512
#define MROWS (BATCH * TSEQ)   // 16384

typedef __attribute__((ext_vector_type(16))) __bf16          v16bf;
typedef __attribute__((ext_vector_type(8)))  float           v8f;
typedef __attribute__((ext_vector_type(8)))  unsigned short  us8;
typedef __attribute__((ext_vector_type(4)))  unsigned int    v4u;
typedef __attribute__((ext_vector_type(8)))  int             v8i;
typedef __attribute__((ext_vector_type(4)))  int             v4i;

union FragU { v16bf v; us8 h[2]; };

// A-matrix 16x32 bf16 fragment (ISA 7.12.2): lane holds row M=lane%16,
// elems 0..7 -> K=base..base+7, elems 8..15 -> K=16+base.., base=(lane>=16)*8.
// Caller passes p = rowptr + kb + base.
__device__ __forceinline__ v16bf load_frag_a(const unsigned short* p) {
  FragU u;
  u.h[0] = *(const us8*)(p);
  u.h[1] = *(const us8*)(p + 16);
  return u.v;
}

// B-matrix 32x16 bf16 fragment: lane holds column N=lane%16,
// elems 0..7 -> K=base..base+7, elems 8..15 -> K=16+base.., base=(lane>=16)*8.
// Caller passes p = &B[base][n] with row stride ld.
__device__ __forceinline__ v16bf load_frag_b(const unsigned short* p, size_t ld) {
  FragU u;
#pragma unroll
  for (int s = 0; s < 8; ++s) {
    u.h[0][s] = p[(size_t)s * ld];
    u.h[1][s] = p[(size_t)(16 + s) * ld];
  }
  return u.v;
}

__device__ __forceinline__ v8f wmma_bf16(v16bf a, v16bf b, v8f c) {
  return __builtin_amdgcn_wmma_f32_16x16x32_bf16(
      false, a, false, b, (short)0, c, false, false);
}

// fp32 -> bf16 bits, round-to-nearest-even
__device__ __forceinline__ unsigned short f2bf(float f) {
  unsigned int b = __float_as_uint(f);
  unsigned int r = b + 0x7FFFu + ((b >> 16) & 1u);
  return (unsigned short)(r >> 16);
}

// ---------------------------------------------------------------------------
// Tensor Data Mover: 2D bf16 tile -> LDS (one wave issues; EXEC-independent).
// D# built per ISA 8.3/8.4; group2/3 zero (2D tensor).
// ---------------------------------------------------------------------------
#if __has_builtin(__builtin_amdgcn_tensor_load_to_lds)
#define HAVE_TDM 1
__device__ __forceinline__ void tdm_load_tile_bf16(
    unsigned int lds_off, const unsigned short* gptr,
    unsigned int tdim0, unsigned int tdim1, unsigned long long stride0,
    unsigned int tile0, unsigned int tile1)
{
  unsigned long long ga = (unsigned long long)(const void*)gptr;
  v4u g0;
  g0[0] = (unsigned)__builtin_amdgcn_readfirstlane(1);          // count=1, user D#
  g0[1] = (unsigned)__builtin_amdgcn_readfirstlane((int)lds_off);
  g0[2] = (unsigned)__builtin_amdgcn_readfirstlane((int)(unsigned)ga);
  g0[3] = (unsigned)__builtin_amdgcn_readfirstlane(
      (int)(((unsigned)(ga >> 32) & 0x01FFFFFFu) | (2u << 30))); // type=2
  v8i g1;
  g1[0] = __builtin_amdgcn_readfirstlane((int)(1u << 16));       // data_size=2B
  g1[1] = __builtin_amdgcn_readfirstlane((int)((tdim0 & 0xFFFFu) << 16));
  g1[2] = __builtin_amdgcn_readfirstlane((int)((tdim0 >> 16) | ((tdim1 & 0xFFFFu) << 16)));
  g1[3] = __builtin_amdgcn_readfirstlane((int)((tdim1 >> 16) | (tile0 << 16)));
  g1[4] = __builtin_amdgcn_readfirstlane((int)(tile1 & 0xFFFFu)); // tile_dim1, dim2=0
  g1[5] = __builtin_amdgcn_readfirstlane((int)(unsigned)(stride0 & 0xFFFFFFFFull));
  g1[6] = __builtin_amdgcn_readfirstlane((int)(unsigned)((stride0 >> 32) & 0xFFFFull));
  g1[7] = 0;
  v4i gz = {0, 0, 0, 0};
#if __clang_major__ >= 23
  v8i gz8 = {0, 0, 0, 0, 0, 0, 0, 0};
  __builtin_amdgcn_tensor_load_to_lds(g0, g1, gz, gz, gz8, 0);
#else
  __builtin_amdgcn_tensor_load_to_lds(g0, g1, gz, gz, 0);
#endif
}
#endif

// ---------------------------------------------------------------------------
// Kernel 1: fused QKV projection, fp32 -> bf16, WMMA GEMM.
// grid.x = 3 * 256 * 8, block = 256. Wave: 16 rows x 32 cols. K transposed out.
// ---------------------------------------------------------------------------
__global__ __launch_bounds__(256) void qkv_proj_kernel(
    const float* __restrict__ toks, const float* __restrict__ Wq,
    const float* __restrict__ Wk,   const float* __restrict__ Wv,
    unsigned short* __restrict__ qw, unsigned short* __restrict__ kw,
    unsigned short* __restrict__ vw)
{
  int bid = blockIdx.x;
  int nt  = bid & 7;   bid >>= 3;
  int mt  = bid & 255; bid >>= 8;
  int mat = bid;                     // 0=q 1=k 2=v
  const float* W = (mat == 0) ? Wq : ((mat == 1) ? Wk : Wv);
  const int m0 = mt * 64, n0 = nt * 64;

  __shared__ __align__(32) unsigned short aT[64 * 32];  // toks tile [row][k]
  __shared__ __align__(32) unsigned short wT[32 * 64];  // W tile transposed [k][n]

  const int tid  = threadIdx.x;
  const int lane = tid & 31;
  const int wave = tid >> 5;
  const int wr   = wave & 3;     // row group (16 rows)
  const int wc   = wave >> 2;    // col group (32 cols)
  const int hi   = lane >> 4;

  const int sr = tid >> 2;             // staging row/col index 0..63
  const int sk = (tid & 3) << 3;       // staging k offset {0,8,16,24}

  v8f acc0 = {}, acc1 = {};

  for (int kb = 0; kb < CDIM; kb += 32) {
    // stage A tile: 8 contiguous elems per thread, vectorized
    {
      const float* src = &toks[(size_t)(m0 + sr) * CDIM + kb + sk];
      float4 f0 = *(const float4*)src;
      float4 f1 = *(const float4*)(src + 4);
      us8 pk;
      pk[0] = f2bf(f0.x); pk[1] = f2bf(f0.y); pk[2] = f2bf(f0.z); pk[3] = f2bf(f0.w);
      pk[4] = f2bf(f1.x); pk[5] = f2bf(f1.y); pk[6] = f2bf(f1.z); pk[7] = f2bf(f1.w);
      *(us8*)&aT[sr * 32 + sk] = pk;
    }
    // stage W^T tile: coalesced b128 reads, transposed b16 stores
    {
      const float* src = &W[(size_t)(n0 + sr) * CDIM + kb + sk];
      float4 f0 = *(const float4*)src;
      float4 f1 = *(const float4*)(src + 4);
      unsigned short wb[8] = { f2bf(f0.x), f2bf(f0.y), f2bf(f0.z), f2bf(f0.w),
                               f2bf(f1.x), f2bf(f1.y), f2bf(f1.z), f2bf(f1.w) };
#pragma unroll
      for (int j = 0; j < 8; ++j) wT[(sk + j) * 64 + sr] = wb[j];
    }
    __syncthreads();

    v16bf afrag = load_frag_a(&aT[(wr * 16 + (lane & 15)) * 32 + (hi << 3)]);
    v16bf b0 = load_frag_b(&wT[(hi << 3) * 64 + wc * 32 + (lane & 15) + 0], 64);
    v16bf b1 = load_frag_b(&wT[(hi << 3) * 64 + wc * 32 + (lane & 15) + 16], 64);
    acc0 = wmma_bf16(afrag, b0, acc0);
    acc1 = wmma_bf16(afrag, b1, acc1);
    __syncthreads();
  }

  const int rbase = m0 + wr * 16 + (hi << 3);
  const int cbase = n0 + wc * 32 + (lane & 15);
  if (mat == 1) {
    // K transposed per batch: kw[b][d][t]
#pragma unroll
    for (int r = 0; r < 8; ++r) {
      int row = rbase + r, bb = row >> 12, tt = row & (TSEQ - 1);
      kw[((size_t)bb * CDIM + cbase) * TSEQ + tt]      = f2bf(acc0[r]);
      kw[((size_t)bb * CDIM + cbase + 16) * TSEQ + tt] = f2bf(acc1[r]);
    }
  } else {
    unsigned short* dst = (mat == 0) ? qw : vw;
#pragma unroll
    for (int r = 0; r < 8; ++r) {
      dst[(size_t)(rbase + r) * CDIM + cbase]      = f2bf(acc0[r]);
      dst[(size_t)(rbase + r) * CDIM + cbase + 16] = f2bf(acc1[r]);
    }
  }
}

// ---------------------------------------------------------------------------
// Kernel 2: causal flash attention. grid.x = B*T/16 = 1024, block = 256.
// Block owns 16 query rows (Q tile TDM'd into LDS once). Key blocks of 128:
// wave w computes S for keys [w*16,w*16+16); wave w owns O d-slice [w*64,+64).
// Softmax stats parallelized 16 threads/row with LDS tree reduction.
// ---------------------------------------------------------------------------
__global__ __launch_bounds__(256) void attn_kernel(
    const unsigned short* __restrict__ qw,   // [B*T][C]   bf16
    const unsigned short* __restrict__ kT,   // [B][C][T]  bf16 (transposed)
    const unsigned short* __restrict__ vw,   // [B*T][C]   bf16
    float* __restrict__ out)                 // [B*T][C]   f32
{
  const float scale = 0.044194173824159216f;   // 512^-0.5

  const int rb = blockIdx.x & 255;
  const int b  = blockIdx.x >> 8;
  const int r0 = rb * 16;

  __shared__ __align__(32) unsigned short Qs[16 * CDIM];   // 16 KB Q tile
  __shared__ __align__(32) float          Sb[16 * 128];
  __shared__ __align__(32) unsigned short Pb[16 * 128];
  __shared__ float red[256];
  __shared__ float mrow[16], lrow[16], srow[16];

  const int tid  = threadIdx.x;
  const int lane = tid & 31;
  const int wave = tid >> 5;
  const int hi   = lane >> 4;
  const int row  = tid >> 4;     // stats row 0..15
  const int seg  = tid & 15;     // stats segment (8 cols)
  const int cb8  = seg << 3;

  const unsigned short* qbase = qw + (size_t)(b * TSEQ + r0) * CDIM;
  const unsigned short* kbase = kT + (size_t)b * CDIM * TSEQ;
  const unsigned short* vbase = vw + (size_t)b * TSEQ * CDIM;

  if (tid < 16) { mrow[tid] = -INFINITY; lrow[tid] = 0.0f; }
#if defined(HAVE_TDM)
  if (wave == 0) {
    tdm_load_tile_bf16((unsigned int)(size_t)(const void*)Qs, qbase,
                       /*tdim0=*/CDIM, /*tdim1=*/16, /*stride0=*/CDIM,
                       /*tile0=*/CDIM, /*tile1=*/16);
    __builtin_amdgcn_s_wait_tensorcnt(0);
  }
#else
  for (int i = tid; i < 16 * CDIM / 8; i += 256)
    *(us8*)&Qs[i * 8] = *(const us8*)(qbase + i * 8);
#endif
  __syncthreads();

  v8f Oacc[4] = {};   // 16 rows x 64 d-slice

  const unsigned short* qrow = &Qs[(lane & 15) * CDIM + (hi << 3)];
  const unsigned short* prow = &Pb[(lane & 15) * 128 + (hi << 3)];

  const int nkey = r0 + 16;   // causal: keys 0 .. r0+15

  for (int jb = 0; jb < nkey; jb += 128) {
    if (jb + 128 < nkey)
      __builtin_prefetch(vbase + (size_t)(jb + 128 + lane) * CDIM, 0, 1);

    // ---- S = Q K^T for this wave's 16-key slice ----
    const int kq0 = jb + wave * 16;
    v8f s = {};
    if (kq0 < nkey) {
#pragma unroll
      for (int db = 0; db < CDIM; db += 32) {
        v16bf a  = load_frag_a(qrow + db);
        v16bf bk = load_frag_b(kbase + (size_t)(db + (hi << 3)) * TSEQ + kq0 + (lane & 15), TSEQ);
        s = wmma_bf16(a, bk, s);
      }
    }
#pragma unroll
    for (int r = 0; r < 8; ++r)
      Sb[(r + (hi << 3)) * 128 + wave * 16 + (lane & 15)] = s[r] * scale;
    __syncthreads();                                       // (1) Sb ready

    // ---- parallel online-softmax stats: 16 threads per row ----
    const int kavail = nkey - jb;
    const int kmax   = (kavail < 128) ? kavail : 128;
    int klim = (r0 + row) - jb;                            // causal inclusive bound
    if (klim > kmax - 1) klim = kmax - 1;

    const float4* sp = (const float4*)&Sb[row * 128 + cb8];
    float4 sa = sp[0], sbv = sp[1];
    float sv[8] = { sa.x, sa.y, sa.z, sa.w, sbv.x, sbv.y, sbv.z, sbv.w };

    float pm = -INFINITY;
#pragma unroll
    for (int j = 0; j < 8; ++j)
      if (cb8 + j <= klim) pm = fmaxf(pm, sv[j]);
    red[tid] = pm;
    __syncthreads();                                       // (2) partial max ready

    if (tid < 16) {
      float bm = -INFINITY;
#pragma unroll
      for (int j = 0; j < 16; ++j) bm = fmaxf(bm, red[tid * 16 + j]);
      float m_old = mrow[tid];
      float m_new = fmaxf(m_old, bm);
      srow[tid] = (m_old == -INFINITY) ? 0.0f : __expf(m_old - m_new);
      mrow[tid] = m_new;
    }
    __syncthreads();                                       // (3) m_new/srow ready

    {
      float mn = mrow[row];
      float ps = 0.0f;
      us8 pk;
#pragma unroll
      for (int j = 0; j < 8; ++j) {
        float p = (cb8 + j <= klim) ? __expf(sv[j] - mn) : 0.0f;
        ps += p;
        pk[j] = f2bf(p);
      }
      *(us8*)&Pb[row * 128 + cb8] = pk;
      red[tid] = ps;
    }
    __syncthreads();                                       // (4) Pb + partial sums

    if (tid < 16) {
      float rs = 0.0f;
#pragma unroll
      for (int j = 0; j < 16; ++j) rs += red[tid * 16 + j];
      lrow[tid] = lrow[tid] * srow[tid] + rs;
    }

    // ---- rescale O and accumulate P V ----
    float scv[8];
#pragma unroll
    for (int r = 0; r < 8; ++r) scv[r] = srow[r + (hi << 3)];
#pragma unroll
    for (int t = 0; t < 4; ++t)
#pragma unroll
      for (int r = 0; r < 8; ++r) Oacc[t][r] *= scv[r];

#pragma unroll
    for (int kk = 0; kk < 128; kk += 32) {
      v16bf ap = load_frag_a(prow + kk);
#pragma unroll
      for (int t = 0; t < 4; ++t) {
        v16bf bv = load_frag_b(
            vbase + (size_t)(jb + kk + (hi << 3)) * CDIM + wave * 64 + t * 16 + (lane & 15), CDIM);
        Oacc[t] = wmma_bf16(ap, bv, Oacc[t]);
      }
    }
    __syncthreads();                                       // (5) protect Sb/Pb/red
  }

  // ---- epilogue: out = O / l ----
  float inv[8];
#pragma unroll
  for (int r = 0; r < 8; ++r) inv[r] = 1.0f / lrow[r + (hi << 3)];

  float* obase = out + (size_t)(b * TSEQ + r0) * CDIM + wave * 64;
#pragma unroll
  for (int t = 0; t < 4; ++t)
#pragma unroll
    for (int r = 0; r < 8; ++r)
      obase[(size_t)(r + (hi << 3)) * CDIM + t * 16 + (lane & 15)] =
          Oacc[t][r] * inv[r];
}

// ---------------------------------------------------------------------------
extern "C" void kernel_launch(void* const* d_in, const int* in_sizes, int n_in,
                              void* d_out, int out_size, void* d_ws, size_t ws_size,
                              hipStream_t stream) {
  const float* toks = (const float*)d_in[0];
  const float* Wq   = (const float*)d_in[1];
  const float* Wk   = (const float*)d_in[2];
  const float* Wv   = (const float*)d_in[3];
  float* out = (float*)d_out;

  unsigned short* qw = (unsigned short*)d_ws;                 // 16 MB
  unsigned short* kw = qw + (size_t)MROWS * CDIM;             // 16 MB
  unsigned short* vw = kw + (size_t)MROWS * CDIM;             // 16 MB

  qkv_proj_kernel<<<3 * (MROWS / 64) * (CDIM / 64), 256, 0, stream>>>(
      toks, Wq, Wk, Wv, qw, kw, vw);
  attn_kernel<<<BATCH * (TSEQ / 16), 256, 0, stream>>>(qw, kw, vw, out);
}
